// _QRNN_49005576848023
// MI455X (gfx1250) — matrix operand order repaired and checked by reading, not compiled, not run on hardware
//
#include <hip/hip_runtime.h>
#include <hip/hip_bf16.h>

typedef __attribute__((ext_vector_type(16))) __bf16 v16bf;
typedef __attribute__((ext_vector_type(8)))  __bf16 v8bf;
typedef __attribute__((ext_vector_type(8)))  float  v8f;

#define T_LEN   2048
#define B_SZ    32
#define D_IN    512
#define H_OUT   512
#define NCH     (2 * B_SZ * H_OUT)          // 32768 scan channels
#define M_TOT   (B_SZ * T_LEN)              // 65536 GEMM rows
#define KSTEPS  (D_IN / 32)                 // 16 K-steps of 32

__device__ __forceinline__ float fast_sigmoid(float x) {
    return 1.0f / (1.0f + __expf(-x));
}
__device__ __forceinline__ float fast_tanh(float x) {
    float e = __expf(-2.0f * x);
    return (1.0f - e) / (1.0f + e);
}

// CDNA5 async copy: global memory -> LDS, tracked by ASYNCcnt.
// loff = per-lane LDS byte offset (low 32 bits of the generic LDS address),
// src  = per-lane global address; 16 bytes per lane.
__device__ __forceinline__ void async_ld_b128(unsigned loff, const __bf16* src) {
    asm volatile("global_load_async_to_lds_b128 %0, %1, off"
                 :: "v"(loff), "v"(src) : "memory");
}
__device__ __forceinline__ void wait_asynccnt0() {
    asm volatile("s_wait_asynccnt 0x0" ::: "memory");
}

// ---------------- prep kernels ----------------

// x fp32 [B*T*D] -> bf16 (same layout). 8 elems / thread.
__global__ void prep_x_kernel(const float* __restrict__ x, __bf16* __restrict__ xb) {
    size_t i = ((size_t)blockIdx.x * 256 + threadIdx.x) * 8;
    float4 v0 = *(const float4*)(x + i);
    float4 v1 = *(const float4*)(x + i + 4);
    v8bf o;
    o[0] = (__bf16)v0.x; o[1] = (__bf16)v0.y; o[2] = (__bf16)v0.z; o[3] = (__bf16)v0.w;
    o[4] = (__bf16)v1.x; o[5] = (__bf16)v1.y; o[6] = (__bf16)v1.z; o[7] = (__bf16)v1.w;
    *(v8bf*)(xb + i) = o;
}

// W fp32 [512][1536] -> bf16 transposed Wt[dir][1536][512] (Wt[dir][n][k] = W[k][n])
__global__ void prep_w_kernel(const float* __restrict__ Wf, const float* __restrict__ Wb,
                              __bf16* __restrict__ wt) {
    int dir = blockIdx.x / 1536;
    int n   = blockIdx.x % 1536;
    const float* W = dir ? Wb : Wf;
    __bf16* dst = wt + ((size_t)dir * 1536 + (size_t)n) * D_IN;
    for (int k = threadIdx.x; k < D_IN; k += 128)
        dst[k] = (__bf16)W[(size_t)k * 1536 + n];
}

// ---------------- fused GEMM + activation ----------------
// grid (M_TOT/128, H_OUT/32, 2), block 128 (4 waves).
// wave -> 2 M-tiles (32 rows) x 2 N-tiles (32 cols) x 3 gates = 12 WMMA accs.
// Each LDS B fragment feeds 2 WMMAs; B tiles async-copied to double-buffered LDS.
// __launch_bounds__(128, 1): lift the occupancy-driven VGPR cap so the 12
// accumulators + pipeline state stay resident (no scratch spills).
__global__ void __launch_bounds__(128, 1)
gemm_gates_kernel(const __bf16* __restrict__ xb,
                  const __bf16* __restrict__ Wt,
                  const float* __restrict__ bias_f,
                  const float* __restrict__ bias_b,
                  float* __restrict__ As,
                  float* __restrict__ Bs,
                  float* __restrict__ Os) {
    __shared__ __bf16 sB[2][6 * 512];   // 2 stages x 6 tiles x 1 KB

    const int lane = threadIdx.x & 31;
    const int wave = threadIdx.x >> 5;
    const int lr   = lane & 15;          // row (A) / col (B,C) within tile
    const int lh   = lane >> 4;          // half-wave selector

    const int m0  = blockIdx.x * 128 + wave * 32;
    const int h0  = blockIdx.y * 32;
    const int dir = blockIdx.z;

    const float* bias = dir ? bias_b : bias_f;
    const __bf16* wt  = Wt + (size_t)dir * 1536 * D_IN;

    // A fragments per ISA 16-bit A 16x32 layout: lanes 0-15 hold K=[0..7]+[16..23],
    // lanes 16-31 hold K=[8..15]+[24..31]
    const __bf16* aLo0 = xb + (size_t)(m0 + lr) * D_IN + lh * 8;
    const __bf16* aLo1 = aLo0 + (size_t)16 * D_IN;

    // Per-thread async stage: 6 tiles x 64 16B-chunks = 384 chunks / 128 thr = 3 each.
    // chunk c: tile tt = c>>6 (g = tt>>1, j = tt&1), u = c&63: lane l = u>>1, half hh = u&1.
    unsigned sbase[2];
    sbase[0] = (unsigned)(unsigned long long)(void*)&sB[0][0];
    sbase[1] = (unsigned)(unsigned long long)(void*)&sB[1][0];

    unsigned ldsoff[3];
    const __bf16* gsrc[3];
#pragma unroll
    for (int i = 0; i < 3; ++i) {
        int c  = threadIdx.x + i * 128;
        int tt = c >> 6;
        int u  = c & 63;
        int l  = u >> 1;
        int hh = u & 1;
        int g  = tt >> 1, j = tt & 1;
        int col = h0 + j * 16 + (l & 15);
        ldsoff[i] = (unsigned)(tt * 1024 + l * 32 + hh * 16);
        gsrc[i]   = wt + ((size_t)(g * H_OUT + col)) * D_IN + (l >> 4) * 8 + hh * 16;
    }

    v8f acc[2][3][2];
#pragma unroll
    for (int mt = 0; mt < 2; ++mt)
#pragma unroll
        for (int g = 0; g < 3; ++g)
#pragma unroll
            for (int j = 0; j < 2; ++j)
                acc[mt][g][j] = (v8f){0.f, 0.f, 0.f, 0.f, 0.f, 0.f, 0.f, 0.f};

    // prologue: stage 0
#pragma unroll
    for (int i = 0; i < 3; ++i)
        async_ld_b128(sbase[0] + ldsoff[i], gsrc[i]);

    for (int ks = 0; ks < KSTEPS; ++ks) {
        wait_asynccnt0();       // this wave's async stores to LDS landed
        __syncthreads();        // everyone's did

        if (ks + 1 < KSTEPS) {  // overlap next stage with this step's WMMAs
            const int kk1 = (ks + 1) * 32;
            const unsigned sb = sbase[(ks + 1) & 1];
#pragma unroll
            for (int i = 0; i < 3; ++i)
                async_ld_b128(sb + ldsoff[i], gsrc[i] + kk1);
        }

        const int kk = ks * 32;
        union { v16bf v; v8bf h[2]; } A0, A1;
        A0.h[0] = *(const v8bf*)(aLo0 + kk);
        A0.h[1] = *(const v8bf*)(aLo0 + kk + 16);
        A1.h[0] = *(const v8bf*)(aLo1 + kk);
        A1.h[1] = *(const v8bf*)(aLo1 + kk + 16);

        const __bf16* sbuf = &sB[ks & 1][0];
#pragma unroll
        for (int g = 0; g < 3; ++g) {
#pragma unroll
            for (int j = 0; j < 2; ++j) {
                v16bf Bm = *(const v16bf*)&sbuf[(g * 2 + j) * 512 + lane * 16];
                acc[0][g][j] = __builtin_amdgcn_wmma_f32_16x16x32_bf16(
                    false, A0.v, false, Bm, (short)0, acc[0][g][j], false, false);
                acc[1][g][j] = __builtin_amdgcn_wmma_f32_16x16x32_bf16(
                    false, A1.v, false, Bm, (short)0, acc[1][g][j], false, false);
            }
        }
    }

    // epilogue: bias + activations, write scan state [T][dir][B][H]
#pragma unroll
    for (int j = 0; j < 2; ++j) {
        const int col = h0 + j * 16 + lr;
        const float bz = bias[col];
        const float bf = bias[H_OUT + col];
        const float bo = bias[2 * H_OUT + col];
#pragma unroll
        for (int mt = 0; mt < 2; ++mt) {
#pragma unroll
            for (int r = 0; r < 8; ++r) {
                const int m = m0 + mt * 16 + r + lh * 8;  // C layout: VGPR r -> M=r / r+8
                const int t = m & (T_LEN - 1);
                const int bi = m >> 11;
                const float z = fast_tanh(acc[mt][0][j][r] + bz);
                const float f = fast_sigmoid(acc[mt][1][j][r] + bf);
                const float o = fast_sigmoid(acc[mt][2][j][r] + bo);
                const size_t idx = (size_t)t * NCH + (size_t)dir * (B_SZ * H_OUT)
                                 + (size_t)bi * H_OUT + col;
                As[idx] = f;
                Bs[idx] = (1.0f - f) * z;
                Os[idx] = o;
            }
        }
    }
}

// ---------------- fo_pool scan + output ----------------
// 1 lane per channel (dir,b,h); coalesced streaming, serial FMA recurrence.
__global__ void fo_scan_kernel(const float* __restrict__ As,
                               const float* __restrict__ Bs,
                               const float* __restrict__ Os,
                               float* __restrict__ out) {
    const int ch  = blockIdx.x * blockDim.x + threadIdx.x;   // 0..32767
    const int dir = ch >> 14;
    const int b   = (ch >> 9) & 31;
    const int h   = ch & 511;
    float c = 0.0f;
    size_t idx = (size_t)ch;
    float* op = out + (size_t)b * T_LEN * 1024 + (size_t)dir * H_OUT + h;
#pragma unroll 4
    for (int t = 0; t < T_LEN; ++t) {
        __builtin_prefetch(&As[idx + 8 * NCH], 0, 1);
        __builtin_prefetch(&Bs[idx + 8 * NCH], 0, 1);
        __builtin_prefetch(&Os[idx + 8 * NCH], 0, 1);
        const float a  = As[idx];
        const float bb = Bs[idx];
        const float o  = Os[idx];
        c = fmaf(a, c, bb);
        op[(size_t)t * 1024] = o * c;
        idx += NCH;
    }
}

// ---------------- launch ----------------
extern "C" void kernel_launch(void* const* d_in, const int* in_sizes, int n_in,
                              void* d_out, int out_size, void* d_ws, size_t ws_size,
                              hipStream_t stream) {
    const float* x   = (const float*)d_in[0];
    const float* W_f = (const float*)d_in[1];
    const float* b_f = (const float*)d_in[2];
    const float* W_b = (const float*)d_in[3];
    const float* b_b = (const float*)d_in[4];
    float* out = (float*)d_out;

    char* ws = (char*)d_ws;
    const size_t XE = (size_t)B_SZ * T_LEN * D_IN;          // 33,554,432 elems
    const size_t X_BYTES  = XE * sizeof(__bf16);            // 64 MB
    const size_t WT_BYTES = (size_t)2 * 1536 * D_IN * sizeof(__bf16); // 3 MB
    const size_t GATE_E   = (size_t)T_LEN * NCH;            // 67,108,864 elems

    __bf16* xb = (__bf16*)ws;
    __bf16* wt = (__bf16*)(ws + X_BYTES);
    float*  As = (float*)(ws + X_BYTES + WT_BYTES);
    float*  Bs = As + GATE_E;
    float*  Os = Bs + GATE_E;

    prep_x_kernel<<<XE / (256 * 8), 256, 0, stream>>>(x, xb);
    prep_w_kernel<<<2 * 1536, 128, 0, stream>>>(W_f, W_b, wt);
    gemm_gates_kernel<<<dim3(M_TOT / 128, H_OUT / 32, 2), 128, 0, stream>>>(
        xb, wt, b_f, b_b, As, Bs, Os);
    fo_scan_kernel<<<NCH / 256, 256, 0, stream>>>(As, Bs, Os, out);
}
